// Cox_Model_62775241998364
// MI455X (gfx1250) — matrix operand order repaired and checked by reading, not compiled, not run on hardware
//
#include <hip/hip_runtime.h>
#include <cstddef>
#include <cstdint>

// ---------------- types ----------------
typedef __bf16 bf16_t;
typedef __attribute__((ext_vector_type(8)))  __bf16 v8bf;
typedef __attribute__((ext_vector_type(16))) __bf16 v16bf;
typedef __attribute__((ext_vector_type(8)))  float  v8f;
typedef unsigned int v4u __attribute__((ext_vector_type(4)));
typedef int v8i_t __attribute__((ext_vector_type(8)));
typedef int v4i_t __attribute__((ext_vector_type(4)));

// ---------------- model dims (compile-time, from reference) ----------------
constexpr int kVOCAB = 32000;
constexpr int kCH = 512,  kCL = 4, kCNH = 8,  kCI = 2048;
constexpr int kVH = 1024, kVL = 6, kVNH = 16, kVI = 4096;
constexpr int kNC = 64, kBSZ = 16, kMV = 64;
constexpr int kNV = kBSZ * kMV;        // 1024 visits
constexpr int kMTOK = kNV * kNC;       // 65536 code tokens
constexpr long long kPRED = (long long)kBSZ * kMV * kVOCAB; // 32,768,000

// LDS tile pitches (elements) - padded for bank spread, 16B aligned rows
constexpr int kAP = 40;   // A tile: 32 cols + 8 pad  -> 80 B/row
constexpr int kBP = 136;  // B tile: 128 cols + 8 pad -> 272 B/row

// =====================================================================
// Tensor Data Mover: 2-D tile load Global -> LDS (CDNA5 ISA ch.8 D#)
//   group0: count=1 | lds_addr | global_addr(57b) | type=2
//   group1: workgroup_mask=0, data_size=1(2B), pad_enable, pad_interval,
//           pad_amount, tensor_dim0/1, tile_dim0/1, tensor_dim0_stride
// dims/strides in element (data_size) units. Groups 2/3 zero (2-D tile).
// =====================================================================
__device__ inline void tdm_load_2d(unsigned lds_off, const void* gaddr,
                                   unsigned tensor_d0, unsigned tensor_d1,
                                   unsigned long long stride0,
                                   unsigned tile_d0, unsigned tile_d1,
                                   unsigned pad_interval, unsigned pad_amount) {
  unsigned long long ga = (unsigned long long)(uintptr_t)gaddr;
  v4u g0;
  g0.x = 1u;                                   // count=1 (valid user D#)
  g0.y = lds_off;                              // LDS byte address
  g0.z = (unsigned)ga;                         // global addr [31:0]
  g0.w = (unsigned)((ga >> 32) & 0x01FFFFFFu)  // global addr [56:32]
         | (2u << 30);                         // type = 2 ("image")
  unsigned w0 = (1u << 16)                     // data_size = 1 -> 2 bytes
              | (1u << 20)                     // pad_enable
              | (pad_interval << 22)
              | (pad_amount << 25);
  unsigned w1 = (tensor_d0 & 0xFFFFu) << 16;                       // dim0[15:0]
  unsigned w2 = ((tensor_d0 >> 16) & 0xFFFFu) | ((tensor_d1 & 0xFFFFu) << 16);
  unsigned w3 = ((tensor_d1 >> 16) & 0xFFFFu) | (tile_d0 << 16);
  unsigned w4 = tile_d1;                                            // tile_dim2=0
  unsigned w5 = (unsigned)stride0;
  unsigned w6 = (unsigned)((stride0 >> 32) & 0xFFFFu);              // dim1 stride=0
  unsigned w7 = 0u;
  v8i_t g1 = { (int)w0, (int)w1, (int)w2, (int)w3,
               (int)w4, (int)w5, (int)w6, (int)w7 };
  v4i_t z4 = { 0, 0, 0, 0 };
  v8i_t z8 = { 0, 0, 0, 0, 0, 0, 0, 0 };
  __builtin_amdgcn_tensor_load_to_lds(g0, g1, z4, z4, z8, 0);
}

// =====================================================================
// Weight convert + transpose: W (N,K) f32 row-major -> Wt (K,N) bf16
// =====================================================================
__global__ __launch_bounds__(256) void convtrans_kernel(
    const float* __restrict__ W, bf16_t* __restrict__ Wt, int N, int K) {
  long long i = (long long)blockIdx.x * blockDim.x + threadIdx.x;
  long long total = (long long)N * K;
  if (i >= total) return;
  int k = (int)(i / N);
  int n = (int)(i % N);
  Wt[i] = (bf16_t)W[(size_t)n * K + k];
}

// =====================================================================
// Embedding gather: codes -> bf16 activations (emb row 0 forced to 0)
// =====================================================================
__global__ __launch_bounds__(256) void embed_kernel(
    const int* __restrict__ codes, const float* __restrict__ emb,
    bf16_t* __restrict__ x) {
  long long i = (long long)blockIdx.x * blockDim.x + threadIdx.x;
  long long total = (long long)kMTOK * kCH;
  if (i >= total) return;
  int c = (int)(i % kCH);
  long long tok = i / kCH;
  int cid = codes[tok];
  float v = (cid == 0) ? 0.0f : emb[(size_t)cid * kCH + c];
  x[i] = (bf16_t)v;
}

// =====================================================================
// WMMA GEMM with TDM double-buffered LDS staging.
// D[M,N] = A[M,K] @ Bt[K,N], bf16 in, f32 accumulate.
// Block: 256 thr = 8 waves; wave tile 32x64; block tile 128x128.
// Wave 0 issues tensor_load_to_lds for the 128x32 A tile and the
// 32x128 B tile of the next k-step; all waves consume via ds_load.
// =====================================================================
__device__ inline v16bf lds_frag_a(const bf16_t* row, int kg) {
  v8bf lo = *(const v8bf*)(row + kg);
  v8bf hi = *(const v8bf*)(row + kg + 16);
  v16bf r;
#pragma unroll
  for (int i = 0; i < 8; ++i) { r[i] = lo[i]; r[i + 8] = hi[i]; }
  return r;
}
__device__ inline v16bf lds_frag_b(const bf16_t* p, int kg) {
  v16bf r;
#pragma unroll
  for (int i = 0; i < 8; ++i) { r[i] = p[(size_t)(kg + i) * kBP]; r[i + 8] = p[(size_t)(kg + 16 + i) * kBP]; }
  return r;
}

template <bool F32OUT>
__global__ __launch_bounds__(256) void gemm_wmma_kernel(
    const bf16_t* __restrict__ A, const bf16_t* __restrict__ Bt,
    void* __restrict__ Dp, const float* __restrict__ bias,
    int M, int N, int K) {
  __shared__ __align__(16) bf16_t Atile[2][128 * kAP];
  __shared__ __align__(16) bf16_t Btile[2][32 * kBP];

  const int lane = threadIdx.x & 31;
  const int w = threadIdx.x >> 5;
  const int wm = w & 3;        // 4 waves along M
  const int wn = w >> 2;       // 2 waves along N
  const int m0 = blockIdx.y * 128 + wm * 32;
  const int n0blk = blockIdx.x * 128;
  const int n0 = n0blk + wn * 64;
  const bool tdmWave = (threadIdx.x < 32);

  const bf16_t* Ag = A + (size_t)blockIdx.y * 128 * K;   // + k0 per step
  const bf16_t* Bg = Bt + n0blk;                         // + k0*N per step

  unsigned ldsA[2] = { (unsigned)(uintptr_t)(void*)&Atile[0][0],
                       (unsigned)(uintptr_t)(void*)&Atile[1][0] };
  unsigned ldsB[2] = { (unsigned)(uintptr_t)(void*)&Btile[0][0],
                       (unsigned)(uintptr_t)(void*)&Btile[1][0] };

  const v8f vzero = {0.f, 0.f, 0.f, 0.f, 0.f, 0.f, 0.f, 0.f};
  v8f acc[2][4];
#pragma unroll
  for (int i = 0; i < 2; ++i)
#pragma unroll
    for (int t = 0; t < 4; ++t) acc[i][t] = vzero;

  // prologue: DMA the k0=0 tiles into buffer 0
  if (tdmWave) {
    tdm_load_2d(ldsA[0], Ag, 32u, 128u, (unsigned long long)K, 32u, 128u, 3u, 3u);
    tdm_load_2d(ldsB[0], Bg, 128u, 32u, (unsigned long long)N, 128u, 32u, 5u, 3u);
  }

  const int kg = (lane >> 4) * 8;
  const int r0 = wm * 32 + (lane & 15);

  int it = 0;
  for (int k0 = 0; k0 < K; k0 += 32, ++it) {
    const int p = it & 1;
    if (tdmWave) __builtin_amdgcn_s_wait_tensorcnt((short)0);
    __syncthreads();  // tile p visible to all waves
    if (tdmWave && (k0 + 32 < K)) {
      const int q = p ^ 1;
      tdm_load_2d(ldsA[q], Ag + (k0 + 32), 32u, 128u,
                  (unsigned long long)K, 32u, 128u, 3u, 3u);
      tdm_load_2d(ldsB[q], Bg + (size_t)(k0 + 32) * N, 128u, 32u,
                  (unsigned long long)N, 128u, 32u, 5u, 3u);
    }

    const bf16_t* Ab = &Atile[p][0];
    const bf16_t* Bb = &Btile[p][0];
    v16bf a0 = lds_frag_a(Ab + (size_t)r0 * kAP, kg);
    v16bf a1 = lds_frag_a(Ab + (size_t)(r0 + 16) * kAP, kg);
    v16bf b0 = lds_frag_b(Bb + (lane & 15) + wn * 64 + 0, kg);
    v16bf b1 = lds_frag_b(Bb + (lane & 15) + wn * 64 + 16, kg);
    v16bf b2 = lds_frag_b(Bb + (lane & 15) + wn * 64 + 32, kg);
    v16bf b3 = lds_frag_b(Bb + (lane & 15) + wn * 64 + 48, kg);

    acc[0][0] = __builtin_amdgcn_wmma_f32_16x16x32_bf16(false, a0, false, b0, (short)0, acc[0][0], false, false);
    acc[1][0] = __builtin_amdgcn_wmma_f32_16x16x32_bf16(false, a1, false, b0, (short)0, acc[1][0], false, false);
    acc[0][1] = __builtin_amdgcn_wmma_f32_16x16x32_bf16(false, a0, false, b1, (short)0, acc[0][1], false, false);
    acc[1][1] = __builtin_amdgcn_wmma_f32_16x16x32_bf16(false, a1, false, b1, (short)0, acc[1][1], false, false);
    acc[0][2] = __builtin_amdgcn_wmma_f32_16x16x32_bf16(false, a0, false, b2, (short)0, acc[0][2], false, false);
    acc[1][2] = __builtin_amdgcn_wmma_f32_16x16x32_bf16(false, a1, false, b2, (short)0, acc[1][2], false, false);
    acc[0][3] = __builtin_amdgcn_wmma_f32_16x16x32_bf16(false, a0, false, b3, (short)0, acc[0][3], false, false);
    acc[1][3] = __builtin_amdgcn_wmma_f32_16x16x32_bf16(false, a1, false, b3, (short)0, acc[1][3], false, false);
  }

  // C/D layout: element v -> m = v + 8*(lane>>4); n = lane&15
  const int nlane = lane & 15;
  const int moff = 8 * (lane >> 4);
#pragma unroll
  for (int i = 0; i < 2; ++i) {
#pragma unroll
    for (int t = 0; t < 4; ++t) {
#pragma unroll
      for (int v = 0; v < 8; ++v) {
        int mm = m0 + i * 16 + moff + v;
        int nn = n0 + t * 16 + nlane;
        if (F32OUT) {
          ((float*)Dp)[(size_t)mm * N + nn] = acc[i][t][v] + bias[nn];
        } else {
          ((bf16_t*)Dp)[(size_t)mm * N + nn] = (bf16_t)acc[i][t][v];
        }
      }
    }
  }
}

// =====================================================================
// Attention (seq=64, hd=64): one block per (batch, head), 64 threads.
// =====================================================================
template <bool CAUSAL>
__global__ __launch_bounds__(64) void attention_kernel(
    const bf16_t* __restrict__ Q, const bf16_t* __restrict__ K,
    const bf16_t* __restrict__ V, bf16_t* __restrict__ O,
    const int* __restrict__ lenptr, int nh, int H) {
  const int b = blockIdx.x / nh;
  const int h = blockIdx.x % nh;
  const int t = threadIdx.x;  // query position
  const int len = lenptr[b];

  __shared__ float Ks[64][65];
  __shared__ float Vs[64][65];
  __shared__ float Sc[64][65];

  const size_t base = (size_t)b * 64 * H + (size_t)h * 64;

  float qreg[64];
  {
    const v8bf* qp = (const v8bf*)(Q + base + (size_t)t * H);
    const v8bf* kp = (const v8bf*)(K + base + (size_t)t * H);
    const v8bf* vp = (const v8bf*)(V + base + (size_t)t * H);
#pragma unroll
    for (int i = 0; i < 8; ++i) {
      v8bf q8 = qp[i], k8 = kp[i], v8 = vp[i];
#pragma unroll
      for (int j = 0; j < 8; ++j) {
        qreg[i * 8 + j] = (float)q8[j];
        Ks[t][i * 8 + j] = (float)k8[j];
        Vs[t][i * 8 + j] = (float)v8[j];
      }
    }
  }
  __syncthreads();

  const float scale = 0.125f;
  for (int k2 = 0; k2 < 64; ++k2) {
    float s = 0.f;
#pragma unroll
    for (int d = 0; d < 64; ++d) s += qreg[d] * Ks[k2][d];
    bool ok = (k2 < len) && (!CAUSAL || (k2 <= t));
    Sc[t][k2] = ok ? s * scale : -3.0e38f;
  }

  float mx = -3.4e38f;
  for (int k2 = 0; k2 < 64; ++k2) mx = fmaxf(mx, Sc[t][k2]);
  float sum = 0.f;
  for (int k2 = 0; k2 < 64; ++k2) {
    float e = __expf(Sc[t][k2] - mx);
    Sc[t][k2] = e;
    sum += e;
  }
  const float inv = 1.0f / sum;
  __syncthreads();

  bf16_t* op = O + base + (size_t)t * H;
  for (int d = 0; d < 64; ++d) {
    float a = 0.f;
    for (int k2 = 0; k2 < 64; ++k2) a += Sc[t][k2] * Vs[k2][d];
    op[d] = (bf16_t)(a * inv);
  }
}

// =====================================================================
// Elementwise helpers
// =====================================================================
__global__ __launch_bounds__(256) void add_kernel(bf16_t* __restrict__ dst,
                                                  const bf16_t* __restrict__ src,
                                                  long long n) {
  long long i = (long long)blockIdx.x * blockDim.x + threadIdx.x;
  if (i < n) dst[i] = (bf16_t)((float)dst[i] + (float)src[i]);
}

__global__ __launch_bounds__(256) void silu_mul_kernel(bf16_t* __restrict__ g,
                                                       const bf16_t* __restrict__ u,
                                                       long long n) {
  long long i = (long long)blockIdx.x * blockDim.x + threadIdx.x;
  if (i < n) {
    float gv = (float)g[i];
    float s = gv / (1.0f + __expf(-gv));
    g[i] = (bf16_t)(s * (float)u[i]);
  }
}

__global__ __launch_bounds__(256) void layernorm_kernel(
    const bf16_t* __restrict__ x, bf16_t* __restrict__ y,
    const float* __restrict__ g, const float* __restrict__ b, int H) {
  const int row = blockIdx.x;
  const int tid = threadIdx.x;
  __shared__ float s1[256];
  __shared__ float s2[256];
  const bf16_t* xr = x + (size_t)row * H;
  float a = 0.f, a2 = 0.f;
  for (int c = tid; c < H; c += 256) {
    float v = (float)xr[c];
    a += v; a2 += v * v;
  }
  s1[tid] = a; s2[tid] = a2;
  __syncthreads();
  for (int s = 128; s > 0; s >>= 1) {
    if (tid < s) { s1[tid] += s1[tid + s]; s2[tid] += s2[tid + s]; }
    __syncthreads();
  }
  const float m = s1[0] / H;
  const float var = s2[0] / H - m * m;
  const float rstd = rsqrtf(var + 1e-5f);
  bf16_t* yr = y + (size_t)row * H;
  for (int c = tid; c < H; c += 256) {
    float v = (float)xr[c];
    yr[c] = (bf16_t)((v - m) * rstd * g[c] + b[c]);
  }
}

__global__ __launch_bounds__(256) void zero_bf16_kernel(bf16_t* p, long long n) {
  long long i = (long long)blockIdx.x * blockDim.x + threadIdx.x;
  if (i < n) p[i] = (bf16_t)0.0f;
}

// =====================================================================
// segment counts / starting offsets / pooling / scatter / output tail
// =====================================================================
__global__ void zero_int_kernel(int* p, int n) {
  int i = blockIdx.x * blockDim.x + threadIdx.x;
  if (i < n) p[i] = 0;
}

__global__ void count_kernel(const int* __restrict__ visit_id, int* plen) {
  int i = blockIdx.x * blockDim.x + threadIdx.x;
  if (i < kNV) atomicAdd(&plen[visit_id[i]], 1);
}

__global__ void starting_kernel(const int* __restrict__ plen, int* starting) {
  if (threadIdx.x == 0 && blockIdx.x == 0) {
    int acc = 0;
    for (int b = 0; b < kBSZ; ++b) { starting[b] = acc; acc += plen[b]; }
  }
}

__global__ __launch_bounds__(256) void pool_scatter_kernel(
    const bf16_t* __restrict__ cv, const int* __restrict__ codes_len,
    const int* __restrict__ visit_id, const int* __restrict__ starting,
    bf16_t* __restrict__ vx) {
  const int v = blockIdx.x;         // 0..kNV-1
  const int len = codes_len[v];
  const int pid = visit_id[v];
  const int fp = pid * kMV + (v - starting[pid]);
  for (int col = threadIdx.x; col < kVH; col += 256) {
    float s = 0.f;
    for (int c = 0; c < len; ++c)
      s += (float)cv[((size_t)v * kNC + c) * kVH + col];
    vx[(size_t)fp * kVH + col] = (bf16_t)(s * (1.0f / kNC));
  }
}

__global__ void plen_out_kernel(const int* __restrict__ plen, float* out) {
  int i = threadIdx.x;
  if (i < kBSZ) out[i] = (float)plen[i];
}

// =====================================================================
// Host orchestration
// =====================================================================
static inline long long ceil_div_ll(long long a, long long b) { return (a + b - 1) / b; }

extern "C" void kernel_launch(void* const* d_in, const int* in_sizes, int n_in,
                              void* d_out, int out_size, void* d_ws, size_t ws_size,
                              hipStream_t stream) {
  (void)in_sizes; (void)n_in; (void)out_size; (void)ws_size;

  const int*   codes     = (const int*)d_in[0];
  const int*   codes_len = (const int*)d_in[1];
  const int*   visit_id  = (const int*)d_in[2];
  const float* emb       = (const float*)d_in[3];
  auto codeL = [&](int l, int w) { return (const float*)d_in[4 + 11 * l + w]; };
  auto visL  = [&](int l, int w) { return (const float*)d_in[48 + 11 * l + w]; };
  const float* Wc2v = (const float*)d_in[114];
  const float* Wh   = (const float*)d_in[115];
  const float* bh   = (const float*)d_in[116];
  float* out = (float*)d_out;

  size_t off = 0;
  auto alloc = [&](size_t bytes) -> void* {
    void* p = (char*)d_ws + off;
    off += (bytes + 255) & ~(size_t)255;
    return p;
  };

  bf16_t* cwT[kCL][7];
  for (int l = 0; l < kCL; ++l) {
    for (int w = 0; w < 4; ++w) cwT[l][w] = (bf16_t*)alloc((size_t)kCH * kCH * 2);
    cwT[l][4] = (bf16_t*)alloc((size_t)kCH * kCI * 2);
    cwT[l][5] = (bf16_t*)alloc((size_t)kCH * kCI * 2);
    cwT[l][6] = (bf16_t*)alloc((size_t)kCI * kCH * 2);
  }
  bf16_t* vwT[kVL][7];
  for (int l = 0; l < kVL; ++l) {
    for (int w = 0; w < 4; ++w) vwT[l][w] = (bf16_t*)alloc((size_t)kVH * kVH * 2);
    vwT[l][4] = (bf16_t*)alloc((size_t)kVH * kVI * 2);
    vwT[l][5] = (bf16_t*)alloc((size_t)kVH * kVI * 2);
    vwT[l][6] = (bf16_t*)alloc((size_t)kVI * kVH * 2);
  }
  bf16_t* c2vT = (bf16_t*)alloc((size_t)kCH * kVH * 2);
  bf16_t* whT  = (bf16_t*)alloc((size_t)kVH * kVOCAB * 2);

  bf16_t* xb = (bf16_t*)alloc((size_t)kMTOK * kCH * 2);
  bf16_t* hb = (bf16_t*)alloc((size_t)kMTOK * kCH * 2);
  bf16_t* qb = (bf16_t*)alloc((size_t)kMTOK * kCH * 2);
  bf16_t* kb = (bf16_t*)alloc((size_t)kMTOK * kCH * 2);
  bf16_t* vb = (bf16_t*)alloc((size_t)kMTOK * kCH * 2);
  bf16_t* ob = (bf16_t*)alloc((size_t)kMTOK * kCH * 2);
  bf16_t* gb = (bf16_t*)alloc((size_t)kMTOK * kCI * 2);  // also reused as cv
  bf16_t* ub = (bf16_t*)alloc((size_t)kMTOK * kCI * 2);

  const int M2 = kBSZ * kMV;  // 1024
  bf16_t* vxb = (bf16_t*)alloc((size_t)M2 * kVH * 2);
  bf16_t* vhb = (bf16_t*)alloc((size_t)M2 * kVH * 2);
  bf16_t* vqb = (bf16_t*)alloc((size_t)M2 * kVH * 2);
  bf16_t* vkb = (bf16_t*)alloc((size_t)M2 * kVH * 2);
  bf16_t* vvb = (bf16_t*)alloc((size_t)M2 * kVH * 2);
  bf16_t* vob = (bf16_t*)alloc((size_t)M2 * kVH * 2);
  bf16_t* vgb = (bf16_t*)alloc((size_t)M2 * kVI * 2);
  bf16_t* vub = (bf16_t*)alloc((size_t)M2 * kVI * 2);

  int* plen  = (int*)alloc(64 * sizeof(int));
  int* start = plen + 16;

  auto convtrans = [&](const float* W, bf16_t* Wt, int N, int K) {
    long long total = (long long)N * K;
    convtrans_kernel<<<(unsigned)ceil_div_ll(total, 256), 256, 0, stream>>>(W, Wt, N, K);
  };
  auto gemm = [&](const bf16_t* A, const bf16_t* Bt, bf16_t* D, int M, int N, int K) {
    dim3 grid(N / 128, M / 128);
    gemm_wmma_kernel<false><<<grid, 256, 0, stream>>>(A, Bt, (void*)D, nullptr, M, N, K);
  };
  auto addv = [&](bf16_t* dst, const bf16_t* src, long long n) {
    add_kernel<<<(unsigned)ceil_div_ll(n, 256), 256, 0, stream>>>(dst, src, n);
  };
  auto lnorm = [&](const bf16_t* x, bf16_t* y, const float* g, const float* b, int M, int H) {
    layernorm_kernel<<<M, 256, 0, stream>>>(x, y, g, b, H);
  };

  // ---- phase 0: convert + transpose all weights to bf16 ----
  for (int l = 0; l < kCL; ++l) {
    for (int w = 0; w < 4; ++w) convtrans(codeL(l, w), cwT[l][w], kCH, kCH);
    convtrans(codeL(l, 4), cwT[l][4], kCI, kCH);
    convtrans(codeL(l, 5), cwT[l][5], kCI, kCH);
    convtrans(codeL(l, 6), cwT[l][6], kCH, kCI);
  }
  for (int l = 0; l < kVL; ++l) {
    for (int w = 0; w < 4; ++w) convtrans(visL(l, w), vwT[l][w], kVH, kVH);
    convtrans(visL(l, 4), vwT[l][4], kVI, kVH);
    convtrans(visL(l, 5), vwT[l][5], kVI, kVH);
    convtrans(visL(l, 6), vwT[l][6], kVH, kVI);
  }
  convtrans(Wc2v, c2vT, kVH, kCH);
  convtrans(Wh, whT, kVOCAB, kVH);

  // ---- phase 1: embedding ----
  {
    long long total = (long long)kMTOK * kCH;
    embed_kernel<<<(unsigned)ceil_div_ll(total, 256), 256, 0, stream>>>(codes, emb, xb);
  }

  // ---- phase 2: code encoder layers ----
  const int M1 = kMTOK;
  for (int l = 0; l < kCL; ++l) {
    gemm(xb, cwT[l][0], qb, M1, kCH, kCH);
    gemm(xb, cwT[l][1], kb, M1, kCH, kCH);
    gemm(xb, cwT[l][2], vb, M1, kCH, kCH);
    attention_kernel<false><<<kNV * kCNH, 64, 0, stream>>>(qb, kb, vb, ob, codes_len, kCNH, kCH);
    gemm(ob, cwT[l][3], hb, M1, kCH, kCH);
    addv(xb, hb, (long long)M1 * kCH);
    lnorm(xb, hb, codeL(l, 7), codeL(l, 8), M1, kCH);
    gemm(hb, cwT[l][4], gb, M1, kCI, kCH);
    gemm(hb, cwT[l][5], ub, M1, kCI, kCH);
    silu_mul_kernel<<<(unsigned)ceil_div_ll((long long)M1 * kCI, 256), 256, 0, stream>>>(
        gb, ub, (long long)M1 * kCI);
    gemm(gb, cwT[l][6], hb, M1, kCH, kCI);
    addv(xb, hb, (long long)M1 * kCH);
    lnorm(xb, xb, codeL(l, 9), codeL(l, 10), M1, kCH);
  }

  // ---- phase 3: project codes to visit space (cv reuses gb) ----
  gemm(xb, c2vT, gb, M1, kVH, kCH);

  // ---- phase 4: patients_len / starting / pool+scatter ----
  zero_int_kernel<<<1, 64, 0, stream>>>(plen, 32);
  count_kernel<<<kNV / 256, 256, 0, stream>>>(visit_id, plen);
  starting_kernel<<<1, 1, 0, stream>>>(plen, start);
  zero_bf16_kernel<<<(unsigned)ceil_div_ll((long long)M2 * kVH, 256), 256, 0, stream>>>(
      vxb, (long long)M2 * kVH);
  pool_scatter_kernel<<<kNV, 256, 0, stream>>>(gb, codes_len, visit_id, start, vxb);

  // ---- phase 5: visit decoder layers (causal) ----
  for (int l = 0; l < kVL; ++l) {
    gemm(vxb, vwT[l][0], vqb, M2, kVH, kVH);
    gemm(vxb, vwT[l][1], vkb, M2, kVH, kVH);
    gemm(vxb, vwT[l][2], vvb, M2, kVH, kVH);
    attention_kernel<true><<<kBSZ * kVNH, 64, 0, stream>>>(vqb, vkb, vvb, vob, plen, kVNH, kVH);
    gemm(vob, vwT[l][3], vhb, M2, kVH, kVH);
    addv(vxb, vhb, (long long)M2 * kVH);
    lnorm(vxb, vhb, visL(l, 7), visL(l, 8), M2, kVH);
    gemm(vhb, vwT[l][4], vgb, M2, kVI, kVH);
    gemm(vhb, vwT[l][5], vub, M2, kVI, kVH);
    silu_mul_kernel<<<(unsigned)ceil_div_ll((long long)M2 * kVI, 256), 256, 0, stream>>>(
        vgb, vub, (long long)M2 * kVI);
    gemm(vgb, vwT[l][6], vhb, M2, kVH, kVI);
    addv(vxb, vhb, (long long)M2 * kVH);
    lnorm(vxb, vxb, visL(l, 9), visL(l, 10), M2, kVH);
  }

  // ---- phase 6: LM head (f32 out + bias) and patients_len tail ----
  {
    dim3 grid(kVOCAB / 128, M2 / 128);
    gemm_wmma_kernel<true><<<grid, 256, 0, stream>>>(vxb, whT, (void*)out, bh, M2, kVOCAB, kVH);
  }
  plen_out_kernel<<<1, 32, 0, stream>>>(plen, out + kPRED);
}
